// FactorizedAttention_49993419325825
// MI455X (gfx1250) — matrix operand order repaired and checked by reading, not compiled, not run on hardware
//
#include <hip/hip_runtime.h>
#include <math.h>

// ===========================================================================
// FactorizedAttention forward for MI455X (gfx1250, wave32, WMMA).
//
// Pipeline (all f16 WMMA w/ f32 accumulation, f32 scalar math elsewhere):
//   prep:   f32->f16 convert of hidden; f32->f16 *transposed* copies of all
//           weight matrices so WMMA B-fragments are contiguous 32B reads.
//   K1/K4:  128x128x32 block-tiled GEMMs, 8 waves/block, DOUBLE-BUFFERED LDS
//           filled with GLOBAL_LOAD_ASYNC_TO_LDS_B128 (ASYNCcnt path): next
//           K-tile streams into LDS while v_wmma_f32_16x16x32_f16 consumes
//           the current one.  64 WMMA per 16KB staged.
//   K2:     per 16-token tile: LayerNorm (VALU f32) -> WMMA low-rank
//           (qn@qw1)*(qn@qw2) -> exact GELU -> WMMA @qproj -> residual.
//   K3:     causal flash attention, 1 wave per 16-query tile, 32-key blocks:
//           4 WMMA for S=Q*K^T, online softmax (shfl_xor butterflies within
//           16-lane halves), P staged through LDS (s_wait_dscnt), 4 WMMA for
//           P*V against pre-transposed V^T.  Never materializes S x S scores
//           (~0.5 GB saved); all live data (~84 MB ws) stays in 192MB L2.
// ===========================================================================

typedef __attribute__((ext_vector_type(16))) _Float16 v16h;
typedef __attribute__((ext_vector_type(8)))  _Float16 v8h;
typedef __attribute__((ext_vector_type(8)))  float    v8f;

#define B_ 2
#define S_ 2048
#define E_ 1024
#define H_ 16
#define D_ 64
#define R_ 128

__device__ __forceinline__ v8f wmma32(v16h a, v16h b, v8f c) {
  // v_wmma_f32_16x16x32_f16  (neg_a, A, neg_b, B, c_mod, C, reuse_a, reuse_b)
  return __builtin_amdgcn_wmma_f32_16x16x32_f16(false, a, false, b, (short)0, c,
                                                false, false);
}

// A-matrix fragment (16x32 f16, row-major source, ld in halves).
// ISA layout: lane L holds row L%16; halves j<8 -> k=(L>=16?8:0)+j,
// halves j>=8 -> k=16+(L>=16?8:0)+(j-8).  Two aligned 16B chunks per lane.
__device__ __forceinline__ v16h frag_a(const _Float16* base, int ld, int k0, int lane) {
  const int row = lane & 15;
  const int h8  = (lane >> 4) << 3;
  const _Float16* p = base + row * ld + k0 + h8;
  v8h lo = *(const v8h*)(p);
  v8h hi = *(const v8h*)(p + 16);
  v16h r;
#pragma unroll
  for (int j = 0; j < 8; ++j) { r[j] = lo[j]; r[8 + j] = hi[j]; }
  return r;
}

// B-matrix fragment (32x16 f16) from an [N][K] row-major source (k contiguous).
// ISA layout: lane L holds col N=L%16; halves j<8 -> k=(L>=16?8:0)+j,
// halves j>=8 -> k=16+(L>=16?8:0)+(j-8) (same k order as frag_a).
__device__ __forceinline__ v16h frag_b(const _Float16* base, int ld, int k0, int lane) {
  const int n   = lane & 15;
  const int h8  = (lane >> 4) << 3;
  const _Float16* p = base + n * ld + k0 + h8;
  v8h lo = *(const v8h*)(p);
  v8h hi = *(const v8h*)(p + 16);
  v16h r;
#pragma unroll
  for (int j = 0; j < 8; ++j) { r[j] = lo[j]; r[8 + j] = hi[j]; }
  return r;
}

// 64B global -> LDS direct async copy (no VGPR staging), ASYNCcnt-tracked.
// Per-lane semantics: each instruction copies 16B from global to LDS, same offset on both sides.
__device__ __forceinline__ void async_copy_row64B(const _Float16* gsrc,
                                                  _Float16* ldst) {
  unsigned lds = (unsigned)(size_t)(void*)ldst;               // low32 = LDS off
  unsigned long long ga = (unsigned long long)(size_t)(const void*)gsrc;
  asm volatile(
      "global_load_async_to_lds_b128 %0, %1, off\n\t"
      "global_load_async_to_lds_b128 %0, %1, off offset:16\n\t"
      "global_load_async_to_lds_b128 %0, %1, off offset:32\n\t"
      "global_load_async_to_lds_b128 %0, %1, off offset:48"
      :: "v"(lds), "v"(ga) : "memory");
}

__device__ __forceinline__ void wait_async_zero() {
  asm volatile("s_wait_asynccnt 0x0" ::: "memory");
}

// ------------------------------- prep kernels ------------------------------

__global__ void cvt_f32_to_f16_kernel(const float* __restrict__ src,
                                      _Float16* __restrict__ dst, int n) {
  int i = blockIdx.x * blockDim.x + threadIdx.x;
  if (i < n) dst[i] = (_Float16)src[i];
}

// src [rows][cols] f32 -> dst [cols][rows] f16
__global__ void transpose_f32_to_f16_kernel(const float* __restrict__ src,
                                            _Float16* __restrict__ dst,
                                            int rows, int cols) {
  int i = blockIdx.x * blockDim.x + threadIdx.x;
  if (i < rows * cols) {
    int r = i / cols, c = i - r * cols;
    dst[(size_t)c * rows + r] = (_Float16)src[i];
  }
}

// --------------- shared 128x128x32 double-buffered GEMM core ---------------
// 256 threads / 8 waves; wave (w&3)->M 32-slab, (w>>2)->N 64-slab.
// acc[mi][nb] covers rows mw+mi*16, cols nw+nb*16.
__device__ __forceinline__ void gemm_core_128x128(
    const _Float16* __restrict__ A, const _Float16* __restrict__ Bt,
    int m0, int n0, _Float16* As0, _Float16* Bs0, _Float16* As1, _Float16* Bs1,
    v8f acc[2][4]) {
  const int tid = threadIdx.x, lane = tid & 31, wv = tid >> 5;
  const int mw = (wv & 3) * 32, nw = (wv >> 2) * 64;
  // each thread owns one 64B row-segment of the (A|B) tile
  const _Float16* grow = (tid < 128) ? (A + (size_t)(m0 + tid) * E_)
                                     : (Bt + (size_t)(n0 + tid - 128) * E_);
  _Float16* l0 = (tid < 128) ? (As0 + tid * 32) : (Bs0 + (tid - 128) * 32);
  _Float16* l1 = (tid < 128) ? (As1 + tid * 32) : (Bs1 + (tid - 128) * 32);

  async_copy_row64B(grow, l0);                       // k-tile 0 -> buf0
  for (int it = 0; it < E_ / 32; ++it) {
    wait_async_zero();                               // tile it resident
    __syncthreads();                                 // ...for ALL waves; and
                                                     // iter it-1 reads done
    if (it + 1 < E_ / 32)                            // stream tile it+1 while
      async_copy_row64B(grow + (it + 1) * 32,        // computing tile it
                        (it & 1) ? l0 : l1);
    const _Float16* Ab = (it & 1) ? As1 : As0;
    const _Float16* Bb = (it & 1) ? Bs1 : Bs0;
    v16h a0 = frag_a(Ab + (mw + 0) * 32, 32, 0, lane);
    v16h a1 = frag_a(Ab + (mw + 16) * 32, 32, 0, lane);
#pragma unroll
    for (int nb = 0; nb < 4; ++nb) {
      v16h b = frag_b(Bb + (nw + nb * 16) * 32, 32, 0, lane);
      acc[0][nb] = wmma32(a0, b, acc[0][nb]);
      acc[1][nb] = wmma32(a1, b, acc[1][nb]);
    }
  }
}

// ------------------------------ K1: qkv GEMM -------------------------------
__global__ __launch_bounds__(256)
void qkv_gemm_kernel(const _Float16* __restrict__ A, const _Float16* __restrict__ Bt,
                     const float* __restrict__ bias,
                     float* __restrict__ q32, float* __restrict__ k32,
                     _Float16* __restrict__ vT) {
  __shared__ __align__(16) _Float16 As[2][128 * 32];
  __shared__ __align__(16) _Float16 Bs[2][128 * 32];
  const int lane = threadIdx.x & 31, wv = threadIdx.x >> 5;
  const int m0 = blockIdx.x * 128, n0 = blockIdx.y * 128;
  v8f acc[2][4] = {};
  gemm_core_128x128(A, Bt, m0, n0, As[0], Bs[0], As[1], Bs[1], acc);

  // epilogue: split into q (f32), k (f32), v (f16 transposed [B,H,D,S])
  const int colb = lane & 15, h8 = (lane >> 4) << 3;
  const int mw = (wv & 3) * 32, nw = (wv >> 2) * 64;
#pragma unroll
  for (int nb = 0; nb < 4; ++nb) {
    const int n = n0 + nw + nb * 16 + colb;      // 0..3071
    const float bz = bias[n];
    const int e = n & (E_ - 1);
    const int which = n >> 10;                   // 0=q 1=k 2=v
    const int h = e >> 6, d = e & 63;
#pragma unroll
    for (int mi = 0; mi < 2; ++mi) {
#pragma unroll
      for (int v = 0; v < 8; ++v) {
        const int m = m0 + mw + mi * 16 + v + h8;  // 0..4095
        const int b = m >> 11, s = m & (S_ - 1);
        const float val = acc[mi][nb][v] + bz;
        const size_t bh = (size_t)b * H_ + h;
        if (which == 0)      q32[(bh * S_ + s) * D_ + d] = val;
        else if (which == 1) k32[(bh * S_ + s) * D_ + d] = val;
        else                 vT[(bh * D_ + d) * S_ + s] = (_Float16)val;
      }
    }
  }
}

// ------------------------------ K4: proj GEMM ------------------------------
__global__ __launch_bounds__(256)
void proj_gemm_kernel(const _Float16* __restrict__ A, const _Float16* __restrict__ Bt,
                      const float* __restrict__ bias, float* __restrict__ out) {
  __shared__ __align__(16) _Float16 As[2][128 * 32];
  __shared__ __align__(16) _Float16 Bs[2][128 * 32];
  const int lane = threadIdx.x & 31, wv = threadIdx.x >> 5;
  const int m0 = blockIdx.x * 128, n0 = blockIdx.y * 128;
  v8f acc[2][4] = {};
  gemm_core_128x128(A, Bt, m0, n0, As[0], Bs[0], As[1], Bs[1], acc);

  const int colb = lane & 15, h8 = (lane >> 4) << 3;
  const int mw = (wv & 3) * 32, nw = (wv >> 2) * 64;
#pragma unroll
  for (int nb = 0; nb < 4; ++nb) {
    const int n = n0 + nw + nb * 16 + colb;
    const float bz = bias[n];
#pragma unroll
    for (int mi = 0; mi < 2; ++mi) {
#pragma unroll
      for (int v = 0; v < 8; ++v) {
        const int m = m0 + mw + mi * 16 + v + h8;
        out[(size_t)m * E_ + n] = acc[mi][nb][v] + bz;
      }
    }
  }
}

// --------------------- K2: LN + low-rank struct path -----------------------
// 1 block = 2 waves; wave0 handles Q, wave1 handles K, for one 16-token tile.
__global__ __launch_bounds__(64)
void struct_kernel(const float* __restrict__ q32, const float* __restrict__ k32,
                   const _Float16* __restrict__ qw1T, const _Float16* __restrict__ qw2T,
                   const _Float16* __restrict__ kw3T, const _Float16* __restrict__ kw4T,
                   const _Float16* __restrict__ qprojT, const _Float16* __restrict__ kprojT,
                   const float* __restrict__ g_q, const float* __restrict__ b_q,
                   const float* __restrict__ g_k, const float* __restrict__ b_k,
                   const float* __restrict__ alpha_q, const float* __restrict__ alpha_k,
                   _Float16* __restrict__ qs, _Float16* __restrict__ ks) {
  __shared__ __align__(16) _Float16 XN[2][16 * D_];   // normalized tile (A layout)
  __shared__ __align__(16) _Float16 G[2][16 * R_];    // gelu(t1*t2) (A layout)
  const int tid = threadIdx.x, lane = tid & 31, wv = tid >> 5;
  const int tile = blockIdx.x;
  const int bh = tile >> 7;                 // 128 q-tiles per (b,h)
  const int s0 = (tile & 127) << 4;
  const int h = bh & (H_ - 1);
  const float* src = (wv ? k32 : q32) + ((size_t)bh * S_ + s0) * D_;
  const _Float16* w1 = wv ? kw3T : qw1T;
  const _Float16* w2 = wv ? kw4T : qw2T;
  const _Float16* wp = wv ? kprojT : qprojT;
  const float* gg = wv ? g_k : g_q;
  const float* bb = wv ? b_k : b_q;
  const float sig = 1.0f / (1.0f + __expf(-((wv ? alpha_k : alpha_q)[h])));
  _Float16* dst = (wv ? ks : qs) + ((size_t)bh * S_ + s0) * D_;

  // LayerNorm over D=64: lane = (row L%16, half-row (L>=16)) -> 32 floats.
  {
    const int r = lane & 15, ch = (lane >> 4) << 5;
    const float* x = src + r * D_ + ch;
    float xs[32], sum = 0.f, sq = 0.f;
#pragma unroll
    for (int j = 0; j < 32; ++j) { xs[j] = x[j]; sum += xs[j]; sq += xs[j] * xs[j]; }
    sum += __shfl_xor(sum, 16, 32);
    sq  += __shfl_xor(sq, 16, 32);
    const float mu = sum * (1.0f / 64.0f);
    const float rstd = rsqrtf(sq * (1.0f / 64.0f) - mu * mu + 1e-5f);
#pragma unroll
    for (int j = 0; j < 32; ++j) {
      const int d = ch + j;
      XN[wv][r * D_ + d] = (_Float16)((xs[j] - mu) * rstd * gg[d] + bb[d]);
    }
  }
  // each wave only touches its own LDS region; per-wave LDS is in-order.
  const v16h a0 = frag_a(XN[wv], D_, 0, lane);
  const v16h a1 = frag_a(XN[wv], D_, 32, lane);
  const int colb = lane & 15, h8 = (lane >> 4) << 3;

  // t1 = xn@w1, t2 = xn@w2  (16x128, K=64), fused exact-GELU(t1*t2) -> G
#pragma unroll
  for (int nb = 0; nb < 8; ++nb) {
    v8f t1 = {}, t2 = {};
    t1 = wmma32(a0, frag_b(w1 + nb * 16 * D_, D_, 0, lane), t1);
    t1 = wmma32(a1, frag_b(w1 + nb * 16 * D_, D_, 32, lane), t1);
    t2 = wmma32(a0, frag_b(w2 + nb * 16 * D_, D_, 0, lane), t2);
    t2 = wmma32(a1, frag_b(w2 + nb * 16 * D_, D_, 32, lane), t2);
#pragma unroll
    for (int v = 0; v < 8; ++v) {
      const float xg = t1[v] * t2[v];
      const float gl = 0.5f * xg * (1.0f + erff(xg * 0.70710678118f));
      G[wv][(v + h8) * R_ + nb * 16 + colb] = (_Float16)gl;
    }
  }
  // raw = G @ proj  (16x64, K=128); struct = x + sigmoid(alpha)*raw -> f16
#pragma unroll
  for (int nb = 0; nb < 4; ++nb) {
    v8f acc = {};
#pragma unroll
    for (int kc = 0; kc < 4; ++kc) {
      v16h a = frag_a(G[wv], R_, kc * 32, lane);
      v16h b = frag_b(wp + nb * 16 * R_, R_, kc * 32, lane);
      acc = wmma32(a, b, acc);
    }
#pragma unroll
    for (int v = 0; v < 8; ++v) {
      const int r = v + h8, d = nb * 16 + colb;
      dst[r * D_ + d] = (_Float16)(src[r * D_ + d] + sig * acc[v]);
    }
  }
}

// --------------------- K3: causal flash attention --------------------------
// 4 waves/block; wave handles one 16-query tile; 32-key blocks; online softmax.
__global__ __launch_bounds__(128)
void attn_kernel(const _Float16* __restrict__ qs, const _Float16* __restrict__ ks,
                 const _Float16* __restrict__ vT, const float* __restrict__ temp,
                 _Float16* __restrict__ ao) {
  __shared__ __align__(16) _Float16 Pl[4][16 * 32];  // per-wave P staging
  const int lane = threadIdx.x & 31, wv = threadIdx.x >> 5;
  const int bh = blockIdx.x, h = bh & (H_ - 1), b = bh >> 4;
  const int q0 = (blockIdx.y * 4 + wv) << 4;
  const _Float16* Q  = qs + ((size_t)bh * S_ + q0) * D_;
  const _Float16* Kp = ks + (size_t)bh * S_ * D_;
  const _Float16* Vp = vT + (size_t)bh * D_ * S_;   // [D][S]
  const float scale = 0.125f / temp[h];             // 1/sqrt(64)/temperature
  const int colb = lane & 15, h8 = (lane >> 4) << 3;

  const v16h aq0 = frag_a(Q, D_, 0, lane);
  const v16h aq1 = frag_a(Q, D_, 32, lane);

  v8f o0 = {}, o1 = {}, o2 = {}, o3 = {};
  float m_i[8], l_i[8];
#pragma unroll
  for (int v = 0; v < 8; ++v) { m_i[v] = -3.0e38f; l_i[v] = 0.f; }
  const int kend = q0 + 16;                         // causal: keys 0..q0+15
  _Float16* P = Pl[wv];

  for (int kb = 0; kb < kend; kb += 32) {
    // S tile (16 queries x 32 keys) = Q * K^T over d=64
    v8f s0 = {}, s1 = {};
    s0 = wmma32(aq0, frag_b(Kp + (size_t)kb * D_, D_, 0, lane), s0);
    s0 = wmma32(aq1, frag_b(Kp + (size_t)kb * D_, D_, 32, lane), s0);
    s1 = wmma32(aq0, frag_b(Kp + (size_t)(kb + 16) * D_, D_, 0, lane), s1);
    s1 = wmma32(aq1, frag_b(Kp + (size_t)(kb + 16) * D_, D_, 32, lane), s1);
#pragma unroll
    for (int v = 0; v < 8; ++v) {
      const int qrow = q0 + v + h8;
      float x0 = s0[v] * scale; if (kb + colb > qrow)      x0 = -3.0e38f;
      float x1 = s1[v] * scale; if (kb + 16 + colb > qrow) x1 = -3.0e38f;
      float mx = fmaxf(x0, x1);                 // row-max butterfly (16 lanes)
      mx = fmaxf(mx, __shfl_xor(mx, 1, 32));
      mx = fmaxf(mx, __shfl_xor(mx, 2, 32));
      mx = fmaxf(mx, __shfl_xor(mx, 4, 32));
      mx = fmaxf(mx, __shfl_xor(mx, 8, 32));
      const float mnew = fmaxf(m_i[v], mx);
      const float p0 = __expf(x0 - mnew), p1 = __expf(x1 - mnew);
      P[(v + h8) * 32 + colb]      = (_Float16)p0;
      P[(v + h8) * 32 + 16 + colb] = (_Float16)p1;
      float rs = p0 + p1;                       // row-sum butterfly
      rs += __shfl_xor(rs, 1, 32);
      rs += __shfl_xor(rs, 2, 32);
      rs += __shfl_xor(rs, 4, 32);
      rs += __shfl_xor(rs, 8, 32);
      const float corr = __expf(m_i[v] - mnew);
      l_i[v] = l_i[v] * corr + rs;
      m_i[v] = mnew;
      o0[v] *= corr; o1[v] *= corr; o2[v] *= corr; o3[v] *= corr;
    }
    asm volatile("s_wait_dscnt 0x0" ::: "memory"); // P stores visible to reads
    const v16h pa = frag_a(P, 32, 0, lane);        // P re-shaped to A layout
    o0 = wmma32(pa, frag_b(Vp + (size_t)0  * S_ + kb, S_, 0, lane), o0);
    o1 = wmma32(pa, frag_b(Vp + (size_t)16 * S_ + kb, S_, 0, lane), o1);
    o2 = wmma32(pa, frag_b(Vp + (size_t)32 * S_ + kb, S_, 0, lane), o2);
    o3 = wmma32(pa, frag_b(Vp + (size_t)48 * S_ + kb, S_, 0, lane), o3);
  }
  // out[b, s, h, d] = o / l   (merged-head layout for the proj GEMM)
#pragma unroll
  for (int v = 0; v < 8; ++v) {
    const int r = q0 + v + h8;
    const float inv = 1.0f / l_i[v];
    const size_t rowbase = ((size_t)(b * S_ + r) * H_ + h) * D_;
    ao[rowbase + 0  + colb] = (_Float16)(o0[v] * inv);
    ao[rowbase + 16 + colb] = (_Float16)(o1[v] * inv);
    ao[rowbase + 32 + colb] = (_Float16)(o2[v] * inv);
    ao[rowbase + 48 + colb] = (_Float16)(o3[v] * inv);
  }
}

// ------------------------------- launcher ----------------------------------
extern "C" void kernel_launch(void* const* d_in, const int* in_sizes, int n_in,
                              void* d_out, int out_size, void* d_ws, size_t ws_size,
                              hipStream_t stream) {
  (void)in_sizes; (void)n_in; (void)out_size; (void)ws_size;
  const float* hs      = (const float*)d_in[0];
  const float* w_attn  = (const float*)d_in[1];
  const float* b_attn  = (const float*)d_in[2];
  const float* w_proj  = (const float*)d_in[3];
  const float* b_proj  = (const float*)d_in[4];
  const float* qw1     = (const float*)d_in[5];
  const float* qw2     = (const float*)d_in[6];
  const float* kw3     = (const float*)d_in[7];
  const float* kw4     = (const float*)d_in[8];
  const float* qproj   = (const float*)d_in[9];
  const float* kproj   = (const float*)d_in[10];
  const float* alpha_q = (const float*)d_in[11];
  const float* alpha_k = (const float*)d_in[12];
  const float* temp    = (const float*)d_in[13];
  const float* g_q     = (const float*)d_in[14];
  const float* beta_q  = (const float*)d_in[15];
  const float* g_k     = (const float*)d_in[16];
  const float* beta_k  = (const float*)d_in[17];
  float* out = (float*)d_out;

  char* w = (char*)d_ws;
  auto take = [&](size_t bytes) {
    char* p = w;
    w += (bytes + 255) & ~(size_t)255;
    return p;
  };
  const size_t NT = (size_t)B_ * S_;               // 4096 tokens
  _Float16* hs16   = (_Float16*)take(NT * E_ * 2);             // 8.4 MB
  _Float16* wattnT = (_Float16*)take((size_t)3 * E_ * E_ * 2); // 6.3 MB
  _Float16* wprojT = (_Float16*)take((size_t)E_ * E_ * 2);     // 2.1 MB
  _Float16* qw1T   = (_Float16*)take((size_t)R_ * D_ * 2);
  _Float16* qw2T   = (_Float16*)take((size_t)R_ * D_ * 2);
  _Float16* kw3T   = (_Float16*)take((size_t)R_ * D_ * 2);
  _Float16* kw4T   = (_Float16*)take((size_t)R_ * D_ * 2);
  _Float16* qprojT = (_Float16*)take((size_t)D_ * R_ * 2);
  _Float16* kprojT = (_Float16*)take((size_t)D_ * R_ * 2);
  float*    q32    = (float*)take(NT * E_ * 4);                // 16.8 MB
  float*    k32    = (float*)take(NT * E_ * 4);                // 16.8 MB
  _Float16* vT16   = (_Float16*)take(NT * E_ * 2);             // 8.4 MB
  _Float16* qs16   = (_Float16*)take(NT * E_ * 2);             // 8.4 MB
  _Float16* ks16   = (_Float16*)take(NT * E_ * 2);             // 8.4 MB
  _Float16* ao16   = (_Float16*)take(NT * E_ * 2);             // 8.4 MB

  // ---- prep: f16 conversions / transposed weight copies ----
  {
    int n = (int)(NT * E_);
    cvt_f32_to_f16_kernel<<<(n + 255) / 256, 256, 0, stream>>>(hs, hs16, n);
  }
  transpose_f32_to_f16_kernel<<<(3 * E_ * E_ + 255) / 256, 256, 0, stream>>>(
      w_attn, wattnT, E_, 3 * E_);
  transpose_f32_to_f16_kernel<<<(E_ * E_ + 255) / 256, 256, 0, stream>>>(
      w_proj, wprojT, E_, E_);
  transpose_f32_to_f16_kernel<<<(D_ * R_ + 255) / 256, 256, 0, stream>>>(qw1, qw1T, D_, R_);
  transpose_f32_to_f16_kernel<<<(D_ * R_ + 255) / 256, 256, 0, stream>>>(qw2, qw2T, D_, R_);
  transpose_f32_to_f16_kernel<<<(D_ * R_ + 255) / 256, 256, 0, stream>>>(kw3, kw3T, D_, R_);
  transpose_f32_to_f16_kernel<<<(D_ * R_ + 255) / 256, 256, 0, stream>>>(kw4, kw4T, D_, R_);
  transpose_f32_to_f16_kernel<<<(R_ * D_ + 255) / 256, 256, 0, stream>>>(qproj, qprojT, R_, D_);
  transpose_f32_to_f16_kernel<<<(R_ * D_ + 255) / 256, 256, 0, stream>>>(kproj, kprojT, R_, D_);

  // ---- K1: qkv GEMM + split/scatter (128x128 tiles, async double-buffer) ----
  qkv_gemm_kernel<<<dim3(NT / 128, (3 * E_) / 128), 256, 0, stream>>>(
      hs16, wattnT, b_attn, q32, k32, vT16);

  // ---- K2: LN + low-rank struct ----
  struct_kernel<<<(unsigned)(B_ * H_ * (S_ / 16)), 64, 0, stream>>>(
      q32, k32, qw1T, qw2T, kw3T, kw4T, qprojT, kprojT,
      g_q, beta_q, g_k, beta_k, alpha_q, alpha_k, qs16, ks16);

  // ---- K3: causal flash attention ----
  attn_kernel<<<dim3(B_ * H_, S_ / 64), 128, 0, stream>>>(
      qs16, ks16, vT16, temp, ao16);

  // ---- K4: output projection (128x128 tiles, async double-buffer) ----
  proj_gemm_kernel<<<dim3(NT / 128, E_ / 128), 256, 0, stream>>>(
      ao16, wprojT, b_proj, out);
}
